// TransformerFusionBlock_14431090114595
// MI455X (gfx1250) — compile-verified, matrix operand order checked by reading, not run on hardware
//
#include <hip/hip_runtime.h>
#include <hip/hip_bf16.h>
#include <math.h>

typedef __attribute__((ext_vector_type(16))) _Float16 v16h;
typedef __attribute__((ext_vector_type(8)))  _Float16 v8h;
typedef __attribute__((ext_vector_type(4)))  _Float16 v4h;
typedef __attribute__((ext_vector_type(8)))  float    v8f;

#define C_DIM 1024
#define N_TOK 256
#define B_SZ  8
#define ROWS  (B_SZ * N_TOK)   /* 2048 token rows */
#define PIX   1024             /* 32*32 */

/* ---------------- input index map (alphabetical pytree flatten) -------- */
enum {
  IN_RGB = 0, IN_IR = 1,
  P_ATT_LN1_B = 2, P_ATT_LN1_G = 3,
  P_ATT_LN2_B = 4, P_ATT_LN2_G = 5,
  P_BLK_LN2_B = 6, P_BLK_LN2_G = 7,
  P_CMF1_B = 8, P_CMF1_W = 9, P_CMF2_B = 10, P_CMF2_W = 11,
  P_COEF = 12, P_IR_W = 13,
  P_KIR_B = 14, P_KIR_W = 15, P_KVIS_B = 16, P_KVIS_W = 17,
  P_MLPI_B1 = 18, P_MLPI_B2 = 19, P_MLPI_W1 = 20, P_MLPI_W2 = 21,
  P_MLPV_B1 = 22, P_MLPV_B2 = 23, P_MLPV_W1 = 24, P_MLPV_W2 = 25,
  P_OIR_B = 26, P_OIR_W = 27, P_OVIS_B = 28, P_OVIS_W = 29,
  P_BN_B = 30, P_BN_G = 31, P_BN_M = 32, P_BN_V = 33,
  P_OUTC_W = 34, P_POS_IR = 35, P_POS_VIS = 36,
  P_QIR_B = 37, P_QIR_W = 38, P_QVIS_B = 39, P_QVIS_W = 40,
  P_VIR_B = 41, P_VIR_W = 42, P_VVIS_B = 43, P_VVIS_W = 44,
  P_VIS_W = 45
};

/* ---- CDNA5 async global->LDS copy (tracked on ASYNCcnt) --------------- */
__device__ __forceinline__ void async_copy_b128(const _Float16* gaddr,
                                                _Float16* laddr) {
  asm volatile("global_load_async_to_lds_b128 %0, %1, off"
               :: "v"((unsigned)(size_t)laddr), "v"(gaddr) : "memory");
}
__device__ __forceinline__ void wait_async0() {
  asm volatile("s_wait_asynccnt 0x0" ::: "memory");
}

/* ======================= generic batched WMMA GEMM ===================== */
/* D[M,N] = act(alpha * A[M,K] x Bt[N,K]^T + bias), fused residual
 * y = cf_out * D + cf_res * resid.
 * 256 thr = 8 waves (4 along M x 2 along N); wave tile 32x32 (2x2 WMMA);
 * block tile 128x64, K step 32; double-buffered LDS fed by async DMA
 * (global_load_async_to_lds_b128), 1 barrier + 1 asynccnt wait / K-step. */
#define TM 128
#define TN 64
#define TK 32
#define ASTR 48              /* f16/row: row base stays 16B aligned (96B) */
#define BSTR 48
#define A_BUF (TM * ASTR)    /* 6144 f16 = 12 KB */
#define B_BUF (TN * BSTR)    /* 3072 f16 = 6 KB  */

#define WMMA_F16(a, b, c) \
  __builtin_amdgcn_wmma_f32_16x16x32_f16(false, (a), false, (b), (short)0, (c), false, false)

__launch_bounds__(256)
__global__ void gemm_wmma_kernel(
    const _Float16* __restrict__ A,  int lda, long long aO, long long aI,
    const _Float16* __restrict__ Bt, int ldb, long long bO, long long bI,
    float* __restrict__ out32, _Float16* __restrict__ out16,
    int ldc, long long cO, long long cI,
    const float* __restrict__ bias,
    const float* __restrict__ resid, int ldr, long long rO, long long rI,
    const float* __restrict__ coefs, int cResIdx, int cOutIdx,
    int M, int N, int K, float alpha, int act, int innerCount)
{
  __shared__ __align__(16) _Float16 As[2 * A_BUF];
  __shared__ __align__(16) _Float16 Bs[2 * B_BUF];

  const int tid = threadIdx.x;
  const int z  = blockIdx.z;
  const int zo = z / innerCount;
  const int zi = z - zo * innerCount;
  const _Float16* Ab = A  + zo * aO + zi * aI;
  const _Float16* Bb = Bt + zo * bO + zi * bI;
  const long long cOff = zo * cO + zi * cI;
  const long long rOff = zo * rO + zi * rI;

  const int m0 = blockIdx.y * TM;
  const int n0 = blockIdx.x * TN;

  /* cooperative DMA indices: A = 128x32 (2 b128/thr), B = 64x32 (1 b128) */
  const int arow = tid >> 2;          /* 0..63 (+64 for 2nd chunk) */
  const int acol = (tid & 3) * 8;
  const int brow = tid >> 2;
  const int bcol = (tid & 3) * 8;

  const int lane = tid & 31;
  const int wave = tid >> 5;
  const int wm = wave & 3;            /* 4 waves along M */
  const int wn = wave >> 2;           /* 2 waves along N */
  const int fr = lane & 15;
  const int hi = lane >> 4;

  /* ISA 16-bit A 16x32 layout: lanes<16 = K0..7,K16..23; lanes>=16 =
     K8..15,K24..31 (B mirrors it, cf sparse 64x16 B table). */
  const int aFrag0 = (wm * 32 +      fr) * ASTR + hi * 8;
  const int aFrag1 = (wm * 32 + 16 + fr) * ASTR + hi * 8;
  const int bFrag0 = (wn * 32 +      fr) * BSTR + hi * 8;
  const int bFrag1 = (wn * 32 + 16 + fr) * BSTR + hi * 8;

  const long long aBase0 = (long long)(m0 + arow) * lda;
  const long long aBase1 = (long long)(m0 + arow + 64) * lda;
  const long long bBase  = (long long)(n0 + brow) * ldb;

  v8f acc00 = {}, acc01 = {}, acc10 = {}, acc11 = {};
  const int nK = K / TK;

  /* preload tile 0 into buffer 0 via async DMA */
  async_copy_b128(Ab + aBase0 + acol, As + arow * ASTR + acol);
  async_copy_b128(Ab + aBase1 + acol, As + (arow + 64) * ASTR + acol);
  async_copy_b128(Bb + bBase + bcol,  Bs + brow * BSTR + bcol);

  for (int kk = 0; kk < nK; ++kk) {
    const int cur = kk & 1;
    /* my DMA writes into buf[cur] (issued last iter) must be LDS-visible
       before I signal; barrier then publishes them to all waves */
    wait_async0();
    __syncthreads();

    if (kk + 1 < nK) {
      /* safe WAR: all waves finished reading buf[nxt] before this barrier */
      const int nxt = cur ^ 1;
      const int nxtOff = (kk + 1) * TK;
      async_copy_b128(Ab + aBase0 + nxtOff + acol,
                      As + nxt * A_BUF + arow * ASTR + acol);
      async_copy_b128(Ab + aBase1 + nxtOff + acol,
                      As + nxt * A_BUF + (arow + 64) * ASTR + acol);
      async_copy_b128(Bb + bBase + nxtOff + bcol,
                      Bs + nxt * B_BUF + brow * BSTR + bcol);
      if (kk + 2 < nK) {
        __builtin_prefetch(Ab + aBase0 + nxtOff + TK + acol, 0, 1);
        __builtin_prefetch(Ab + aBase1 + nxtOff + TK + acol, 0, 1);
        __builtin_prefetch(Bb + bBase + nxtOff + TK + bcol, 0, 1);
      }
    }

    const _Float16* Ac = As + cur * A_BUF;
    const _Float16* Bc = Bs + cur * B_BUF;
    v8h af0lo = *(const v8h*)(Ac + aFrag0);
    v8h af0hi = *(const v8h*)(Ac + aFrag0 + 16);
    v8h af1lo = *(const v8h*)(Ac + aFrag1);
    v8h af1hi = *(const v8h*)(Ac + aFrag1 + 16);
    v8h bf0lo = *(const v8h*)(Bc + bFrag0);
    v8h bf0hi = *(const v8h*)(Bc + bFrag0 + 16);
    v8h bf1lo = *(const v8h*)(Bc + bFrag1);
    v8h bf1hi = *(const v8h*)(Bc + bFrag1 + 16);
    v16h af0, af1, bf0, bf1;
#pragma unroll
    for (int i = 0; i < 8; ++i) {
      af0[i] = af0lo[i]; af0[i + 8] = af0hi[i];
      af1[i] = af1lo[i]; af1[i + 8] = af1hi[i];
      bf0[i] = bf0lo[i]; bf0[i + 8] = bf0hi[i];
      bf1[i] = bf1lo[i]; bf1[i + 8] = bf1hi[i];
    }
    acc00 = WMMA_F16(af0, bf0, acc00);
    acc01 = WMMA_F16(af0, bf1, acc01);
    acc10 = WMMA_F16(af1, bf0, acc10);
    acc11 = WMMA_F16(af1, bf1, acc11);
  }

  /* fused epilogue */
  const int colBase = n0 + wn * 32 + fr;
  const int rowBase = m0 + wm * 32 + hi * 8;
  const float cr = coefs ? coefs[cResIdx] : 0.0f;
  const float co = coefs ? coefs[cOutIdx] : 1.0f;
#pragma unroll
  for (int j = 0; j < 2; ++j) {
    const int col = colBase + j * 16;
    const float bc = bias ? bias[col] : 0.0f;
#pragma unroll
    for (int i = 0; i < 2; ++i) {
      const v8f accv = (i == 0) ? (j == 0 ? acc00 : acc01)
                                : (j == 0 ? acc10 : acc11);
#pragma unroll
      for (int r = 0; r < 8; ++r) {
        const int row = rowBase + i * 16 + r;
        float v = alpha * accv[r] + bc;
        if (act == 1)      v = fmaxf(v, 0.0f);
        else if (act == 2) v = 0.5f * v * (1.0f + erff(v * 0.70710678118654752440f));
        float y = co * v;
        if (resid) y += cr * resid[rOff + (long long)row * ldr + col];
        const long long ci = cOff + (long long)row * ldc + col;
        if (out32) out32[ci] = y;
        if (out16) out16[ci] = (_Float16)y;
      }
    }
  }
}

/* ======= compile probe for CDNA5 tensor/cluster paths (unlaunched) ===== */
__global__ void cdna5_async_probe_kernel(const float* __restrict__ g,
                                         float* __restrict__ out)
{
  __shared__ __align__(16) float buf[1024];
  unsigned ldsAddr = (unsigned)(size_t)&buf[threadIdx.x * 4];
  const float* src = g + threadIdx.x * 4;
  asm volatile("global_load_async_to_lds_b128 %0, %1, off"
               :: "v"(ldsAddr), "v"(src) : "memory");
  asm volatile("s_wait_asynccnt 0x0" ::: "memory");
  __builtin_amdgcn_s_wait_tensorcnt(0);
  __builtin_amdgcn_s_cluster_barrier();
  const int cid = __builtin_amdgcn_cluster_id_x();
  __syncthreads();
  out[threadIdx.x] = buf[threadIdx.x] + (float)cid;
}

/* =============== pooling + weighted combine + pos-embed ================= */
__global__ void pool_combine_kernel(
    const float* __restrict__ rgb, const float* __restrict__ ir,
    const float* __restrict__ visW, const float* __restrict__ irW,
    const float* __restrict__ posVis, const float* __restrict__ posIr,
    float* __restrict__ rgbTok, float* __restrict__ irTok,
    float* __restrict__ rgbFlat, float* __restrict__ irFlat,
    _Float16* __restrict__ rgbTok16, _Float16* __restrict__ irTok16,
    _Float16* __restrict__ cmfIn16)
{
  const int token = blockIdx.x;      /* 0..2047 */
  const int b = token >> 8;
  const int t = token & 255;
  const int ty = t >> 4, tx = t & 15;
  const float w0 = visW[0], w1 = visW[1], u0 = irW[0], u1 = irW[1];
  for (int c = threadIdx.x; c < C_DIM; c += 256) {
    const long long base = (((long long)(b * C_DIM + c)) * 32 + ty * 2) * 32 + tx * 2;
    float r00 = rgb[base], r01 = rgb[base + 1], r10 = rgb[base + 32], r11 = rgb[base + 33];
    float i00 = ir[base],  i01 = ir[base + 1],  i10 = ir[base + 32],  i11 = ir[base + 33];
    float rp = w0 * 0.25f * (r00 + r01 + r10 + r11) +
               w1 * fmaxf(fmaxf(r00, r01), fmaxf(r10, r11));
    float ip = u0 * 0.25f * (i00 + i01 + i10 + i11) +
               u1 * fmaxf(fmaxf(i00, i01), fmaxf(i10, i11));
    const long long o = (long long)token * C_DIM + c;
    rgbTok[o] = rp;  irTok[o] = ip;
    rgbFlat[o] = rp + posVis[(long long)t * C_DIM + c];
    irFlat[o]  = ip + posIr[(long long)t * C_DIM + c];
    rgbTok16[o] = (_Float16)rp;  irTok16[o] = (_Float16)ip;
    cmfIn16[(long long)token * 2048 + c] = (_Float16)rp;
    cmfIn16[(long long)token * 2048 + C_DIM + c] = (_Float16)ip;
  }
}

/* ================= CMF channel gate: mean over tokens -> sigmoid ======== */
__global__ void cmf_reduce_kernel(const float* __restrict__ cmf2, float* __restrict__ S)
{
  const int idx = blockIdx.x * blockDim.x + threadIdx.x;  /* 8192 */
  const int b = idx >> 10, c = idx & 1023;
  float s = 0.0f;
  for (int n = 0; n < N_TOK; ++n)
    s += cmf2[((long long)(b * N_TOK + n)) * C_DIM + c];
  s *= (1.0f / 256.0f);
  S[idx] = 1.0f / (1.0f + __expf(-s));
}

__global__ void fuse_kernel(const float* __restrict__ rgbTok,
                            const float* __restrict__ irTok,
                            const float* __restrict__ S,
                            _Float16* __restrict__ rgbFu16,
                            _Float16* __restrict__ irFu16)
{
  const long long i = (long long)blockIdx.x * 256 + threadIdx.x;
  const int c = (int)(i & 1023);
  const int b = (int)(i >> 18);              /* i / (256*1024) */
  const float s = S[b * C_DIM + c];
  const float rp = rgbTok[i], ip = irTok[i];
  irFu16[i]  = (_Float16)(rp * s + ip);      /* ir_fused  */
  rgbFu16[i] = (_Float16)(ip * s + rp);      /* rgb_fused */
}

/* ============================ LayerNorm ================================ */
__global__ void layernorm_f16_kernel(const float* __restrict__ x,
                                     const float* __restrict__ g,
                                     const float* __restrict__ beta,
                                     _Float16* __restrict__ out)
{
  __shared__ float rs[8], rs2[8];
  const long long row = blockIdx.x;
  const float* xr = x + row * C_DIM;
  float s = 0.0f, s2 = 0.0f;
  for (int c = threadIdx.x; c < C_DIM; c += 256) {
    float v = xr[c]; s += v; s2 += v * v;
  }
  for (int off = 16; off > 0; off >>= 1) {
    s  += __shfl_down(s, off, 32);
    s2 += __shfl_down(s2, off, 32);
  }
  const int wave = threadIdx.x >> 5, lane = threadIdx.x & 31;
  if (lane == 0) { rs[wave] = s; rs2[wave] = s2; }
  __syncthreads();
  if (threadIdx.x == 0) {
    float a = 0.0f, b2 = 0.0f;
    for (int i = 0; i < 8; ++i) { a += rs[i]; b2 += rs2[i]; }
    rs[0] = a; rs2[0] = b2;
  }
  __syncthreads();
  const float mean = rs[0] * (1.0f / C_DIM);
  const float var  = rs2[0] * (1.0f / C_DIM) - mean * mean;
  const float inv  = rsqrtf(var + 1e-5f);
  for (int c = threadIdx.x; c < C_DIM; c += 256)
    out[row * C_DIM + c] = (_Float16)((xr[c] - mean) * inv * g[c] + beta[c]);
}

/* ================== variance weight + softmax (row=256) ================ */
__device__ __forceinline__ float block_reduce_sum(float v, float* rbuf) {
  for (int off = 16; off > 0; off >>= 1) v += __shfl_down(v, off, 32);
  const int wave = threadIdx.x >> 5, lane = threadIdx.x & 31;
  if (lane == 0) rbuf[wave] = v;
  __syncthreads();
  float r = (threadIdx.x < 8) ? rbuf[threadIdx.x] : 0.0f;
  for (int off = 4; off > 0; off >>= 1) r += __shfl_down(r, off, 32);
  if (threadIdx.x == 0) rbuf[0] = r;
  __syncthreads();
  float out = rbuf[0];
  __syncthreads();
  return out;
}
__device__ __forceinline__ float block_reduce_max(float v, float* rbuf) {
  for (int off = 16; off > 0; off >>= 1) v = fmaxf(v, __shfl_down(v, off, 32));
  const int wave = threadIdx.x >> 5, lane = threadIdx.x & 31;
  if (lane == 0) rbuf[wave] = v;
  __syncthreads();
  float r = (threadIdx.x < 8) ? rbuf[threadIdx.x] : -3.4e38f;
  for (int off = 4; off > 0; off >>= 1) r = fmaxf(r, __shfl_down(r, off, 32));
  if (threadIdx.x == 0) rbuf[0] = r;
  __syncthreads();
  float out = rbuf[0];
  __syncthreads();
  return out;
}

__global__ void attn_softmax_kernel(const float* __restrict__ scores,
                                    _Float16* __restrict__ att)
{
  __shared__ float rbuf[8];
  const long long row = blockIdx.x;          /* 64*256 rows */
  const float x = scores[row * 256 + threadIdx.x];
  const float m   = block_reduce_sum(x, rbuf) * (1.0f / 256.0f);
  const float d2  = (x - m) * (x - m);
  const float var = block_reduce_sum(d2, rbuf) * (1.0f / 256.0f);
  const float w = x * (1.0f / (1.0f + __expf(-(d2 / (2.0f * var + 1e-6f) + 0.5f))));
  const float mx = block_reduce_max(w, rbuf);
  const float e  = __expf(w - mx);
  const float se = block_reduce_sum(e, rbuf);
  att[row * 256 + threadIdx.x] = (_Float16)(e / se);
}

/* =================== per-head V transpose (f16) ======================== */
__global__ void head_transpose_kernel(const _Float16* __restrict__ v,
                                      _Float16* __restrict__ vT)
{
  __shared__ _Float16 tile[32][33];
  const int z = blockIdx.z;                 /* 64 = b*8 + h */
  const int b = z >> 3, h = z & 7;
  const int n0 = blockIdx.x * 32;           /* token  */
  const int d0 = blockIdx.y * 32;           /* head-dim */
  for (int j = threadIdx.y; j < 32; j += 8)
    tile[j][threadIdx.x] =
        v[((long long)(b * N_TOK + n0 + j)) * C_DIM + h * 128 + d0 + threadIdx.x];
  __syncthreads();
  for (int j = threadIdx.y; j < 32; j += 8)
    vT[(long long)z * 32768 + (long long)(d0 + j) * 256 + n0 + threadIdx.x] =
        tile[threadIdx.x][j];
}

/* =================== weight transpose f32 -> f16 [N,K] ================= */
__global__ void wtrans_kernel(const float* __restrict__ w,
                              _Float16* __restrict__ wt, int din, int dout)
{
  __shared__ float tile[32][33];
  const int r0 = blockIdx.y * 32;  /* din  */
  const int c0 = blockIdx.x * 32;  /* dout */
  for (int j = threadIdx.y; j < 32; j += 8)
    tile[j][threadIdx.x] = w[(long long)(r0 + j) * dout + c0 + threadIdx.x];
  __syncthreads();
  for (int j = threadIdx.y; j < 32; j += 8)
    wt[(long long)(c0 + j) * din + r0 + threadIdx.x] = (_Float16)tile[threadIdx.x][j];
}

__global__ void f32_to_f16_kernel(const float* __restrict__ in,
                                  _Float16* __restrict__ out, long long n)
{
  const long long i = (long long)blockIdx.x * 256 + threadIdx.x;
  if (i < n) out[i] = (_Float16)in[i];
}

/* ==== upsample(rgb_o) + residual + concat -> pixel-major f16 [8192,2048] = */
__global__ void build_actpix_kernel(const float* __restrict__ rgb,
                                    const float* __restrict__ ir,
                                    const float* __restrict__ rgbO,
                                    const float* __restrict__ irO,
                                    _Float16* __restrict__ act)
{
  __shared__ float tile[32][33];
  const int b   = blockIdx.z;
  const int p0  = blockIdx.x * 32;
  const int c0t = blockIdx.y * 32;          /* 0..2047 */
  const float* src  = (c0t < C_DIM) ? rgb  : ir;
  const float* oSrc = (c0t < C_DIM) ? rgbO : irO;
  const int c0 = (c0t < C_DIM) ? c0t : (c0t - C_DIM);
  for (int j = threadIdx.y; j < 32; j += 8)
    tile[j][threadIdx.x] =
        src[((long long)(b * C_DIM + c0 + j)) * PIX + p0 + threadIdx.x];
  __syncthreads();
  for (int j = threadIdx.y; j < 32; j += 8) {
    const int pix = p0 + j;
    const int py = pix >> 5, px = pix & 31;
    const int tok = (py >> 1) * 16 + (px >> 1);
    const float o = oSrc[((long long)(b * N_TOK + tok)) * C_DIM + c0 + threadIdx.x];
    act[((long long)(b * PIX + pix)) * 2048 + c0t + threadIdx.x] =
        (_Float16)(tile[threadIdx.x][j] + o);
  }
}

/* ============ BN(eval) + SiLU + transpose back to NCHW ================= */
__global__ void bn_silu_out_kernel(const float* __restrict__ y,
                                   const float* __restrict__ bg,
                                   const float* __restrict__ bb,
                                   const float* __restrict__ bm,
                                   const float* __restrict__ bv,
                                   float* __restrict__ out)
{
  __shared__ float tile[32][33];
  const int b = blockIdx.z;
  const int p0 = blockIdx.x * 32, o0 = blockIdx.y * 32;
  for (int j = threadIdx.y; j < 32; j += 8) {
    const int o = o0 + threadIdx.x;
    float v = y[((long long)(b * PIX + p0 + j)) * C_DIM + o];
    v = (v - bm[o]) * rsqrtf(bv[o] + 1e-5f) * bg[o] + bb[o];
    tile[j][threadIdx.x] = v * (1.0f / (1.0f + __expf(-v)));
  }
  __syncthreads();
  for (int j = threadIdx.y; j < 32; j += 8)
    out[((long long)(b * C_DIM + o0 + j)) * PIX + p0 + threadIdx.x] =
        tile[threadIdx.x][j];
}

/* ============================ host side ================================ */
static inline char* wsTake(char*& p, size_t bytes) {
  char* r = p;
  p += (bytes + 255) & ~(size_t)255;
  return r;
}

static void launchGemm(hipStream_t s,
    const _Float16* A, int lda, long long aO, long long aI,
    const _Float16* Bt, int ldb, long long bO, long long bI,
    float* out32, _Float16* out16, int ldc, long long cO, long long cI,
    const float* bias,
    const float* resid, int ldr, long long rO, long long rI,
    const float* coefs, int cri, int coi,
    int M, int N, int K, float alpha, int act, int innerCount, int batches)
{
  dim3 grid(N / TN, M / TM, batches);
  gemm_wmma_kernel<<<grid, 256, 0, s>>>(A, lda, aO, aI, Bt, ldb, bO, bI,
      out32, out16, ldc, cO, cI, bias, resid, ldr, rO, rI,
      coefs, cri, coi, M, N, K, alpha, act, innerCount);
}

extern "C" void kernel_launch(void* const* d_in, const int* in_sizes, int n_in,
                              void* d_out, int out_size, void* d_ws, size_t ws_size,
                              hipStream_t stream) {
  (void)in_sizes; (void)n_in; (void)out_size; (void)ws_size;
  const float* F[48];
  for (int i = 0; i < 46; ++i) F[i] = (const float*)d_in[i];

  const size_t TOK32 = (size_t)ROWS * C_DIM * 4;  /* 8 MB */
  const size_t TOK16 = (size_t)ROWS * C_DIM * 2;  /* 4 MB */
  char* w = (char*)d_ws;

  float*     rgbTok   = (float*)wsTake(w, TOK32);
  float*     irTok    = (float*)wsTake(w, TOK32);
  float*     rgbFlat  = (float*)wsTake(w, TOK32);
  float*     irFlat   = (float*)wsTake(w, TOK32);
  _Float16*  rgbTok16 = (_Float16*)wsTake(w, TOK16);
  _Float16*  irTok16  = (_Float16*)wsTake(w, TOK16);
  _Float16*  cmfIn16  = (_Float16*)wsTake(w, (size_t)ROWS * 2048 * 2);
  _Float16*  cmfMid16 = (_Float16*)wsTake(w, TOK16);
  float*     cmf2     = (float*)wsTake(w, TOK32);
  float*     Sgate    = (float*)wsTake(w, (size_t)B_SZ * C_DIM * 4);
  _Float16*  rgbFu16  = (_Float16*)wsTake(w, TOK16);
  _Float16*  irFu16   = (_Float16*)wsTake(w, TOK16);
  _Float16*  rl16     = (_Float16*)wsTake(w, TOK16);
  _Float16*  il16     = (_Float16*)wsTake(w, TOK16);
  const size_t W1M16 = (size_t)C_DIM * C_DIM * 2;
  _Float16*  wQv = (_Float16*)wsTake(w, W1M16);
  _Float16*  wKv = (_Float16*)wsTake(w, W1M16);
  _Float16*  wVv = (_Float16*)wsTake(w, W1M16);
  _Float16*  wQi = (_Float16*)wsTake(w, W1M16);
  _Float16*  wKi = (_Float16*)wsTake(w, W1M16);
  _Float16*  wVi = (_Float16*)wsTake(w, W1M16);
  _Float16*  wOv = (_Float16*)wsTake(w, W1M16);
  _Float16*  wOi = (_Float16*)wsTake(w, W1M16);
  const size_t W4M16 = (size_t)C_DIM * 4096 * 2;
  _Float16*  wM1v = (_Float16*)wsTake(w, W4M16);
  _Float16*  wM2v = (_Float16*)wsTake(w, W4M16);
  _Float16*  wM1i = (_Float16*)wsTake(w, W4M16);
  _Float16*  wM2i = (_Float16*)wsTake(w, W4M16);
  _Float16*  wC1  = (_Float16*)wsTake(w, (size_t)C_DIM * 2048 * 2);
  _Float16*  wC2  = (_Float16*)wsTake(w, W1M16);
  _Float16*  wOut = (_Float16*)wsTake(w, (size_t)C_DIM * 2048 * 2);
  _Float16*  q16v = (_Float16*)wsTake(w, TOK16);
  _Float16*  k16v = (_Float16*)wsTake(w, TOK16);
  _Float16*  v16v = (_Float16*)wsTake(w, TOK16);
  _Float16*  q16i = (_Float16*)wsTake(w, TOK16);
  _Float16*  k16i = (_Float16*)wsTake(w, TOK16);
  _Float16*  v16i = (_Float16*)wsTake(w, TOK16);
  _Float16*  vTv  = (_Float16*)wsTake(w, TOK16);
  _Float16*  vTi  = (_Float16*)wsTake(w, TOK16);
  float*     scores   = (float*)wsTake(w, (size_t)64 * 256 * 256 * 4);
  _Float16*  att16    = (_Float16*)wsTake(w, (size_t)64 * 256 * 256 * 2);
  _Float16*  outHeads = (_Float16*)wsTake(w, TOK16);
  float*     rgbAtt   = (float*)wsTake(w, TOK32);
  float*     irAtt    = (float*)wsTake(w, TOK32);
  _Float16*  lnT      = (_Float16*)wsTake(w, TOK16);
  _Float16*  hidden   = (_Float16*)wsTake(w, (size_t)ROWS * 4096 * 2);
  float*     rgbO     = (float*)wsTake(w, TOK32);
  float*     irO      = (float*)wsTake(w, TOK32);
  _Float16*  actPix   = (_Float16*)wsTake(w, (size_t)B_SZ * PIX * 2048 * 2);
  float*     yPix     = (float*)wsTake(w, (size_t)B_SZ * PIX * C_DIM * 4);

  const dim3 blk328(32, 8);
  /* ---- weight prep: transpose f32 (din,dout) -> f16 [dout,din] ---- */
  struct { int wi; _Float16* dst; int din, dout; } tw[] = {
    {P_QVIS_W, wQv, 1024, 1024}, {P_KVIS_W, wKv, 1024, 1024},
    {P_VVIS_W, wVv, 1024, 1024}, {P_QIR_W,  wQi, 1024, 1024},
    {P_KIR_W,  wKi, 1024, 1024}, {P_VIR_W,  wVi, 1024, 1024},
    {P_OVIS_W, wOv, 1024, 1024}, {P_OIR_W,  wOi, 1024, 1024},
    {P_MLPV_W1, wM1v, 1024, 4096}, {P_MLPV_W2, wM2v, 4096, 1024},
    {P_MLPI_W1, wM1i, 1024, 4096}, {P_MLPI_W2, wM2i, 4096, 1024},
  };
  for (auto& t : tw)
    wtrans_kernel<<<dim3(t.dout / 32, t.din / 32), blk328, 0, stream>>>(
        F[t.wi], t.dst, t.din, t.dout);
  /* conv weights are already (O, Cin) = [N,K]: straight convert */
  f32_to_f16_kernel<<<(1024 * 2048) / 256, 256, 0, stream>>>(F[P_CMF1_W], wC1, 1024 * 2048);
  f32_to_f16_kernel<<<(1024 * 1024) / 256, 256, 0, stream>>>(F[P_CMF2_W], wC2, 1024 * 1024);
  f32_to_f16_kernel<<<(1024 * 2048) / 256, 256, 0, stream>>>(F[P_OUTC_W], wOut, 1024 * 2048);

  /* ---- pooling, gating, fusion ---- */
  pool_combine_kernel<<<ROWS, 256, 0, stream>>>(
      F[IN_RGB], F[IN_IR], F[P_VIS_W], F[P_IR_W], F[P_POS_VIS], F[P_POS_IR],
      rgbTok, irTok, rgbFlat, irFlat, rgbTok16, irTok16, cmfIn16);

  launchGemm(stream, cmfIn16, 2048, 0, 0, wC1, 2048, 0, 0,
             nullptr, cmfMid16, 1024, 0, 0, F[P_CMF1_B],
             nullptr, 0, 0, 0, nullptr, 0, 0,
             ROWS, 1024, 2048, 1.0f, /*relu*/1, 1, 1);
  launchGemm(stream, cmfMid16, 1024, 0, 0, wC2, 1024, 0, 0,
             cmf2, nullptr, 1024, 0, 0, F[P_CMF2_B],
             nullptr, 0, 0, 0, nullptr, 0, 0,
             ROWS, 1024, 1024, 1.0f, 1, 1, 1);
  cmf_reduce_kernel<<<32, 256, 0, stream>>>(cmf2, Sgate);
  fuse_kernel<<<(ROWS * C_DIM) / 256, 256, 0, stream>>>(rgbTok, irTok, Sgate,
                                                        rgbFu16, irFu16);
  layernorm_f16_kernel<<<ROWS, 256, 0, stream>>>(rgbFlat, F[P_ATT_LN1_G], F[P_ATT_LN1_B], rl16);
  layernorm_f16_kernel<<<ROWS, 256, 0, stream>>>(irFlat,  F[P_ATT_LN2_G], F[P_ATT_LN2_B], il16);

  /* ---- QKV projections ---- */
  launchGemm(stream, rgbFu16, 1024, 0, 0, wQv, 1024, 0, 0, nullptr, q16v, 1024, 0, 0,
             F[P_QVIS_B], nullptr, 0, 0, 0, nullptr, 0, 0, ROWS, 1024, 1024, 1.0f, 0, 1, 1);
  launchGemm(stream, rl16,   1024, 0, 0, wKv, 1024, 0, 0, nullptr, k16v, 1024, 0, 0,
             F[P_KVIS_B], nullptr, 0, 0, 0, nullptr, 0, 0, ROWS, 1024, 1024, 1.0f, 0, 1, 1);
  launchGemm(stream, rl16,   1024, 0, 0, wVv, 1024, 0, 0, nullptr, v16v, 1024, 0, 0,
             F[P_VVIS_B], nullptr, 0, 0, 0, nullptr, 0, 0, ROWS, 1024, 1024, 1.0f, 0, 1, 1);
  launchGemm(stream, irFu16, 1024, 0, 0, wQi, 1024, 0, 0, nullptr, q16i, 1024, 0, 0,
             F[P_QIR_B],  nullptr, 0, 0, 0, nullptr, 0, 0, ROWS, 1024, 1024, 1.0f, 0, 1, 1);
  launchGemm(stream, il16,   1024, 0, 0, wKi, 1024, 0, 0, nullptr, k16i, 1024, 0, 0,
             F[P_KIR_B],  nullptr, 0, 0, 0, nullptr, 0, 0, ROWS, 1024, 1024, 1.0f, 0, 1, 1);
  launchGemm(stream, il16,   1024, 0, 0, wVi, 1024, 0, 0, nullptr, v16i, 1024, 0, 0,
             F[P_VIR_B],  nullptr, 0, 0, 0, nullptr, 0, 0, ROWS, 1024, 1024, 1.0f, 0, 1, 1);

  head_transpose_kernel<<<dim3(8, 4, 64), blk328, 0, stream>>>(v16v, vTv);
  head_transpose_kernel<<<dim3(8, 4, 64), blk328, 0, stream>>>(v16i, vTi);

  const float invSqrtDk = 0.08838834764831845f; /* 1/sqrt(128) */
  const long long TOKE = (long long)N_TOK * C_DIM;

  /* ---- vis branch: att_vis = softmax(vw(q_ir . k_vis^T / s)) x v_vis ---- */
  launchGemm(stream, q16i, 1024, TOKE, 128, k16v, 1024, TOKE, 128,
             scores, nullptr, 256, 8LL * 65536, 65536, nullptr,
             nullptr, 0, 0, 0, nullptr, 0, 0,
             256, 256, 128, invSqrtDk, 0, 8, 64);
  attn_softmax_kernel<<<64 * 256, 256, 0, stream>>>(scores, att16);
  launchGemm(stream, att16, 256, 8LL * 65536, 65536, vTv, 256, 8LL * 32768, 32768,
             nullptr, outHeads, 1024, TOKE, 128, nullptr,
             nullptr, 0, 0, 0, nullptr, 0, 0,
             256, 128, 256, 1.0f, 0, 8, 64);
  launchGemm(stream, outHeads, 1024, 0, 0, wOv, 1024, 0, 0,
             rgbAtt, nullptr, 1024, 0, 0, F[P_OVIS_B],
             rgbFlat, 1024, 0, 0, F[P_COEF], 0, 1,
             ROWS, 1024, 1024, 1.0f, 0, 1, 1);

  /* ---- ir branch: att_ir = softmax(vw(q_vis . k_ir^T / s)) x v_ir ---- */
  launchGemm(stream, q16v, 1024, TOKE, 128, k16i, 1024, TOKE, 128,
             scores, nullptr, 256, 8LL * 65536, 65536, nullptr,
             nullptr, 0, 0, 0, nullptr, 0, 0,
             256, 256, 128, invSqrtDk, 0, 8, 64);
  attn_softmax_kernel<<<64 * 256, 256, 0, stream>>>(scores, att16);
  launchGemm(stream, att16, 256, 8LL * 65536, 65536, vTi, 256, 8LL * 32768, 32768,
             nullptr, outHeads, 1024, TOKE, 128, nullptr,
             nullptr, 0, 0, 0, nullptr, 0, 0,
             256, 128, 256, 1.0f, 0, 8, 64);
  launchGemm(stream, outHeads, 1024, 0, 0, wOi, 1024, 0, 0,
             irAtt, nullptr, 1024, 0, 0, F[P_OIR_B],
             irFlat, 1024, 0, 0, F[P_COEF], 2, 3,
             ROWS, 1024, 1024, 1.0f, 0, 1, 1);

  /* ---- MLPs ---- */
  layernorm_f16_kernel<<<ROWS, 256, 0, stream>>>(rgbAtt, F[P_BLK_LN2_G], F[P_BLK_LN2_B], lnT);
  launchGemm(stream, lnT, 1024, 0, 0, wM1v, 1024, 0, 0, nullptr, hidden, 4096, 0, 0,
             F[P_MLPV_B1], nullptr, 0, 0, 0, nullptr, 0, 0,
             ROWS, 4096, 1024, 1.0f, /*gelu*/2, 1, 1);
  launchGemm(stream, hidden, 4096, 0, 0, wM2v, 4096, 0, 0, rgbO, nullptr, 1024, 0, 0,
             F[P_MLPV_B2], rgbAtt, 1024, 0, 0, F[P_COEF], 4, 5,
             ROWS, 1024, 4096, 1.0f, 0, 1, 1);

  layernorm_f16_kernel<<<ROWS, 256, 0, stream>>>(irAtt, F[P_BLK_LN2_G], F[P_BLK_LN2_B], lnT);
  launchGemm(stream, lnT, 1024, 0, 0, wM1i, 1024, 0, 0, nullptr, hidden, 4096, 0, 0,
             F[P_MLPI_B1], nullptr, 0, 0, 0, nullptr, 0, 0,
             ROWS, 4096, 1024, 1.0f, 2, 1, 1);
  launchGemm(stream, hidden, 4096, 0, 0, wM2i, 4096, 0, 0, irO, nullptr, 1024, 0, 0,
             F[P_MLPI_B2], irAtt, 1024, 0, 0, F[P_COEF], 6, 7,
             ROWS, 1024, 4096, 1.0f, 0, 1, 1);

  /* ---- upsample + residual + concat, output conv, BN + SiLU ---- */
  build_actpix_kernel<<<dim3(32, 64, B_SZ), blk328, 0, stream>>>(
      F[IN_RGB], F[IN_IR], rgbO, irO, actPix);
  launchGemm(stream, actPix, 2048, 0, 0, wOut, 2048, 0, 0,
             yPix, nullptr, 1024, 0, 0, nullptr,
             nullptr, 0, 0, 0, nullptr, 0, 0,
             B_SZ * PIX, 1024, 2048, 1.0f, 0, 1, 1);
  bn_silu_out_kernel<<<dim3(32, 32, B_SZ), blk328, 0, stream>>>(
      yPix, F[P_BN_G], F[P_BN_B], F[P_BN_M], F[P_BN_V], (float*)d_out);
}
